// GraphLSTM_90005334655303
// MI455X (gfx1250) — compile-verified, matrix operand order checked by reading, not run on hardware
//
#include <hip/hip_runtime.h>
#include <math.h>

// ---------------- problem constants (from reference) ----------------
#define N_NODES   262144   // B*N
#define N_EDGES   1048576  // B*E
#define N_SAMP    65536    // B*S

typedef __attribute__((ext_vector_type(16))) _Float16 v16h;
typedef __attribute__((ext_vector_type(2)))  _Float16 v2h;
typedef __attribute__((ext_vector_type(8)))  float    v8f;

// ---------------- WMMA fragment tiles in LDS ----------------
// A 16(M)x32(K) or B 32(K)x16(N) tile stored *fragment-major*: lane-l's 16 f16
// elements contiguous at tile[l*16 .. l*16+15] (32 B, aligned).  ISA 7.12.2 map:
//   lane = (mn & 15) + 16*k[3]
//   elem = (k & 7) | ((k & 16) >> 1)     (elem pairs (2p,2p+1) <-> VGPR p)
// Staging writes a (k even, k+1) pair as one b32; loads are one aligned v16h
// (= 2 x ds_load_b128).
__device__ __forceinline__ void frag_store2(_Float16* tile, int mn, int kloc,
                                            float v0, float v1) {
  const int lane = (mn & 15) + ((kloc & 8) << 1);
  const int elem = (kloc & 7) | ((kloc & 16) >> 1);
  v2h p;
  p[0] = (_Float16)v0;
  p[1] = (_Float16)v1;
  *(v2h*)(tile + lane * 16 + elem) = p;
}

__device__ __forceinline__ v16h frag_load(const _Float16* tile) {
  return *(const v16h*)(tile + (threadIdx.x & 31) * 16);
}

__device__ __forceinline__ v8f wmma16(v16h a, v16h b, v8f c) {
  return __builtin_amdgcn_wmma_f32_16x16x32_f16(false, a, false, b, (short)0, c,
                                                false, false);
}

__device__ __forceinline__ float sigf(float v) { return 1.0f / (1.0f + __expf(-v)); }

__device__ __forceinline__ float edge_feat(const float* __restrict__ x,
                                           const float* __restrict__ ea,
                                           int srcN, int dstN, long long e, int k) {
  if (k < 26) return x[srcN * 26 + k];
  if (k < 52) return x[dstN * 26 + (k - 26)];
  if (k < 67) return ea[e * 15 + (k - 52)];
  return 0.0f;
}

__device__ __forceinline__ float node_feat(const float* __restrict__ x,
                                           const float* __restrict__ agg,
                                           int node, int k) {
  if (k < 26) return x[node * 26 + k];
  if (k < 90) return agg[(long long)node * 64 + (k - 26)];
  return 0.0f;
}

// ================= Kernel 1: edge messages + atomic segment-sum =================
// m = ReLU([x[src]|x[dst]|ea] @ W_msg + b) ; agg[dst] += m.  Wave tile: 16 edges x 64.
__global__ __launch_bounds__(256) void edge_msg_kernel(
    const float* __restrict__ x, const int* __restrict__ ei,
    const float* __restrict__ ea, const float* __restrict__ Wmsg,
    const float* __restrict__ bmsg, float* __restrict__ agg) {
  __shared__ __align__(32) _Float16 sW[3][4][512];   // K=96 (3 chunks) x N=64 (4 tiles)
  __shared__ __align__(32) _Float16 sA[8][3][512];   // per-wave 16x96
  __shared__ int sDst[8][16];
  const int tid = threadIdx.x, lane = tid & 31, w = tid >> 5;

  for (int i = tid; i < 48 * 64; i += 256) {
    const int n = i & 63, k = (i >> 6) * 2;
    const float v0 = (k < 67) ? Wmsg[k * 64 + n] : 0.0f;
    const float v1 = (k + 1 < 67) ? Wmsg[(k + 1) * 64 + n] : 0.0f;
    frag_store2(sW[k >> 5][n >> 4], n, k & 31, v0, v1);
  }
  const int ebase = blockIdx.x * 128 + w * 16;
  const int row = lane & 15;
  const long long e = ebase + row;
  const int srcN = ei[e];
  const int dstN = ei[N_EDGES + e];
  if (lane < 16) sDst[w][lane] = dstN;
  const int khalf = (lane >> 4) * 48;
#pragma unroll 4
  for (int k2 = 0; k2 < 48; k2 += 2) {
    const int k = khalf + k2;
    frag_store2(sA[w][k >> 5], row, k & 31,
                edge_feat(x, ea, srcN, dstN, e, k),
                edge_feat(x, ea, srcN, dstN, e, k + 1));
  }
  __syncthreads();

  v8f acc[4] = {};
#pragma unroll
  for (int kt = 0; kt < 3; ++kt) {
    const v16h a = frag_load(sA[w][kt]);
#pragma unroll
    for (int nt = 0; nt < 4; ++nt)
      acc[nt] = wmma16(a, frag_load(sW[kt][nt]), acc[nt]);
  }
  const int n0 = lane & 15, r0 = (lane >> 4) << 3;
#pragma unroll
  for (int nt = 0; nt < 4; ++nt) {
    const int col = nt * 16 + n0;
    const float bias = bmsg[col];
#pragma unroll
    for (int r = 0; r < 8; ++r) {
      const float v = acc[nt][r] + bias;
      if (v > 0.0f)
        atomicAdd(&agg[(long long)sDst[w][r0 + r] * 64 + col], v);
    }
  }
}

// ================= Kernel 2: node update =================
// all_conv = ReLU([x|agg] @ W_node + b), K=90->96.
__global__ __launch_bounds__(256) void node_conv_kernel(
    const float* __restrict__ x, const float* __restrict__ agg,
    const float* __restrict__ Wnode, const float* __restrict__ bnode,
    float* __restrict__ all_conv) {
  __shared__ __align__(32) _Float16 sW[3][4][512];
  __shared__ __align__(32) _Float16 sA[8][3][512];
  const int tid = threadIdx.x, lane = tid & 31, w = tid >> 5;

  for (int i = tid; i < 48 * 64; i += 256) {
    const int n = i & 63, k = (i >> 6) * 2;
    const float v0 = (k < 90) ? Wnode[k * 64 + n] : 0.0f;
    const float v1 = (k + 1 < 90) ? Wnode[(k + 1) * 64 + n] : 0.0f;
    frag_store2(sW[k >> 5][n >> 4], n, k & 31, v0, v1);
  }
  const int nbase = blockIdx.x * 128 + w * 16;
  const int row = lane & 15;
  const int node = nbase + row;
  const int khalf = (lane >> 4) * 48;
#pragma unroll 4
  for (int k2 = 0; k2 < 48; k2 += 2) {
    const int k = khalf + k2;
    frag_store2(sA[w][k >> 5], row, k & 31,
                node_feat(x, agg, node, k), node_feat(x, agg, node, k + 1));
  }
  __syncthreads();

  v8f acc[4] = {};
#pragma unroll
  for (int kt = 0; kt < 3; ++kt) {
    const v16h a = frag_load(sA[w][kt]);
#pragma unroll
    for (int nt = 0; nt < 4; ++nt)
      acc[nt] = wmma16(a, frag_load(sW[kt][nt]), acc[nt]);
  }
  const int n0 = lane & 15, r0 = (lane >> 4) << 3;
#pragma unroll
  for (int nt = 0; nt < 4; ++nt) {
    const int col = nt * 16 + n0;
    const float bias = bnode[col];
#pragma unroll
    for (int r = 0; r < 8; ++r)
      all_conv[(long long)(nbase + r0 + r) * 64 + col] = fmaxf(acc[nt][r] + bias, 0.0f);
  }
}

// ================= Kernel 3: gather + encoder (X and H0) =================
// Waves 0-3: H0 = ReLU(nc @ W_enc + b) + emit node_conv output.
// Waves 4-7: X from gm-substituted rows.  K=64, N=128.
__global__ __launch_bounds__(256) void encoder_kernel(
    const float* __restrict__ all_conv, const int* __restrict__ sidx,
    const float* __restrict__ gm, const float* __restrict__ Wenc,
    const float* __restrict__ benc, float* __restrict__ out_nc,
    _Float16* __restrict__ Xws, _Float16* __restrict__ Hws) {
  __shared__ __align__(32) _Float16 sWe[2][8][512];
  __shared__ __align__(32) _Float16 sAh[4][2][512];
  __shared__ __align__(32) _Float16 sAx[4][2][512];
  const int tid = threadIdx.x, lane = tid & 31, w = tid >> 5;

  for (int i = tid; i < 32 * 128; i += 256) {
    const int n = i & 127, k = (i >> 7) * 2;
    frag_store2(sWe[k >> 5][n >> 4], n, k & 31,
                Wenc[k * 128 + n], Wenc[(k + 1) * 128 + n]);
  }
  const int mt = w & 3;
  const int sbase = blockIdx.x * 64 + mt * 16;
  const int row = lane & 15;
  const int s = sbase + row;
  const int node = sidx[s];
  const int khalf = (lane >> 4) * 32;
  if (w < 4) {
#pragma unroll 4
    for (int k2 = 0; k2 < 32; k2 += 2) {
      const int k = khalf + k2;
      const float v0 = all_conv[(long long)node * 64 + k];
      const float v1 = all_conv[(long long)node * 64 + k + 1];
      frag_store2(sAh[mt][k >> 5], row, k & 31, v0, v1);
      out_nc[(long long)s * 64 + k] = v0;       // output 0: node_conv
      out_nc[(long long)s * 64 + k + 1] = v1;
    }
  } else {
    const int b10 = (s >> 10) * 10;
#pragma unroll 4
    for (int k2 = 0; k2 < 32; k2 += 2) {
      const int k = khalf + k2;
      const float v0 = (k >= 54) ? gm[b10 + k - 54]
                                 : all_conv[(long long)node * 64 + k];
      const float v1 = (k + 1 >= 54) ? gm[b10 + k + 1 - 54]
                                     : all_conv[(long long)node * 64 + k + 1];
      frag_store2(sAx[mt][k >> 5], row, k & 31, v0, v1);
    }
  }
  __syncthreads();

  v8f acc[8] = {};
#pragma unroll
  for (int kt = 0; kt < 2; ++kt) {
    const v16h a = frag_load((w < 4) ? sAh[mt][kt] : sAx[mt][kt]);
#pragma unroll
    for (int nt = 0; nt < 8; ++nt)
      acc[nt] = wmma16(a, frag_load(sWe[kt][nt]), acc[nt]);
  }
  _Float16* outp = (w < 4) ? Hws : Xws;
  const int n0 = lane & 15, r0 = (lane >> 4) << 3;
#pragma unroll
  for (int nt = 0; nt < 8; ++nt) {
    const int col = nt * 16 + n0;
    const float bias = benc[col];
#pragma unroll
    for (int r = 0; r < 8; ++r)
      outp[(long long)(sbase + r0 + r) * 128 + col] =
          (_Float16)fmaxf(acc[nt][r] + bias, 0.0f);
  }
}

// ================= Kernel 4: fused LSTM cell =================
// g = [Xin|H0] @ [W_ih;W_hh] + b_ih + b_hh  (K=256, N=512); gates in-register.
// Wave w: row-tile (w&1), 32-column slice (w>>1) of every gate segment.
__global__ __launch_bounds__(256) void lstm_kernel(
    const void* __restrict__ xin, int xin_f16, int subst,
    const _Float16* __restrict__ H0, const float* __restrict__ gm,
    const float* __restrict__ Wih, const float* __restrict__ Whh,
    const float* __restrict__ bih, const float* __restrict__ bhh,
    float* __restrict__ h_out, float* __restrict__ c_out) {
  __shared__ __align__(32) _Float16 sA[2][8][512];   // 32 rows x 256
  __shared__ __align__(32) _Float16 sB[32][512];     // one K=32 chunk, N=512
  const int tid = threadIdx.x, lane = tid & 31, w = tid >> 5;
  const int mt = w & 1, colq = w >> 1;
  const int sbase = blockIdx.x * 32;

  for (int i = tid; i < 32 * 128; i += 256) {
    const int rrow = i >> 7, k = (i & 127) * 2;   // pairs: gm boundary 118 is even,
    const long long s = sbase + rrow;             // Xin/H0 boundary 128 is even.
    float v0, v1;
    if (k < 128) {
      if (subst && k >= 118) {
        const int b10 = ((int)s >> 10) * 10;
        v0 = gm[b10 + (k - 118)];
        v1 = gm[b10 + (k - 117)];
      } else if (xin_f16) {
        v0 = (float)((const _Float16*)xin)[s * 128 + k];
        v1 = (float)((const _Float16*)xin)[s * 128 + k + 1];
      } else {
        v0 = ((const float*)xin)[s * 128 + k];
        v1 = ((const float*)xin)[s * 128 + k + 1];
      }
    } else {
      v0 = (float)H0[s * 128 + (k - 128)];
      v1 = (float)H0[s * 128 + (k - 127)];
    }
    frag_store2(sA[rrow >> 4][k >> 5], rrow & 15, k & 31, v0, v1);
  }

  v8f acc[4][2] = {};
  for (int kt = 0; kt < 8; ++kt) {
    __syncthreads();
    const float* Wsrc = (kt < 4) ? (Wih + kt * 32 * 512) : (Whh + (kt - 4) * 32 * 512);
    for (int i = tid; i < 16 * 512; i += 256) {
      const int n = i & 511, kk = (i >> 9) * 2;
      frag_store2(sB[n >> 4], n, kk, Wsrc[kk * 512 + n], Wsrc[(kk + 1) * 512 + n]);
    }
    __syncthreads();
    const v16h a = frag_load(sA[mt][kt]);
#pragma unroll
    for (int g = 0; g < 4; ++g)
#pragma unroll
      for (int t = 0; t < 2; ++t)
        acc[g][t] = wmma16(a, frag_load(sB[g * 8 + colq * 2 + t]), acc[g][t]);
  }

  const int n0 = lane & 15, r0 = (lane >> 4) << 3;
#pragma unroll
  for (int t = 0; t < 2; ++t) {
    const int col = colq * 32 + t * 16 + n0;
    const float bi = bih[col]       + bhh[col];
    const float bf = bih[128 + col] + bhh[128 + col];
    const float bg = bih[256 + col] + bhh[256 + col];
    const float bo = bih[384 + col] + bhh[384 + col];
#pragma unroll
    for (int r = 0; r < 8; ++r) {
      const long long s = sbase + mt * 16 + r0 + r;
      const float gi = acc[0][t][r] + bi;
      const float gf = acc[1][t][r] + bf;
      const float gg = acc[2][t][r] + bg;
      const float go = acc[3][t][r] + bo;
      const float c0 = (float)H0[s * 128 + col];   // c_in = H0 in the reference
      const float c2 = sigf(gf) * c0 + sigf(gi) * tanhf(gg);
      const float h2 = sigf(go) * tanhf(c2);
      c_out[s * 128 + col] = c2;
      h_out[s * 128 + col] = h2;
    }
  }
}

// ================= Kernel 5: decoder =================
// d1 = ReLU([h2|c2] @ W_d1 + b) (K=256,N=64 via WMMA); y = d1 @ W_d2 + b (64x6 scalar).
__global__ __launch_bounds__(128) void decoder_kernel(
    const float* __restrict__ h2, const float* __restrict__ c2,
    const float* __restrict__ Wd1, const float* __restrict__ bd1,
    const float* __restrict__ Wd2, const float* __restrict__ bd2,
    float* __restrict__ y) {
  __shared__ __align__(32) _Float16 sW[8][4][512];
  __shared__ __align__(32) _Float16 sA[2][8][512];
  __shared__ float sD[2][16 * 64];
  const int tid = threadIdx.x, lane = tid & 31, w = tid >> 5;
  const int mt = w & 1, nh = w >> 1;
  const int sbase0 = blockIdx.x * 32;

  for (int i = tid; i < 128 * 64; i += 128) {
    const int n = i & 63, k = (i >> 6) * 2;
    frag_store2(sW[k >> 5][n >> 4], n, k & 31, Wd1[k * 64 + n], Wd1[(k + 1) * 64 + n]);
  }
  for (int i = tid; i < 32 * 128; i += 128) {
    const int rrow = i >> 7, k = (i & 127) * 2;
    const long long s = sbase0 + rrow;
    const float* src = (k < 128) ? &h2[s * 128 + k] : &c2[s * 128 + (k - 128)];
    frag_store2(sA[rrow >> 4][k >> 5], rrow & 15, k & 31, src[0], src[1]);
  }
  __syncthreads();

  v8f acc[2] = {};
#pragma unroll
  for (int kt = 0; kt < 8; ++kt) {
    const v16h a = frag_load(sA[mt][kt]);
#pragma unroll
    for (int t = 0; t < 2; ++t)
      acc[t] = wmma16(a, frag_load(sW[kt][nh * 2 + t]), acc[t]);
  }
  const int n0 = lane & 15, r0 = (lane >> 4) << 3;
#pragma unroll
  for (int t = 0; t < 2; ++t) {
    const int col = (nh * 2 + t) * 16 + n0;
    const float bias = bd1[col];
#pragma unroll
    for (int r = 0; r < 8; ++r)
      sD[mt][(r0 + r) * 64 + col] = fmaxf(acc[t][r] + bias, 0.0f);
  }
  __syncthreads();

  for (int t = 0; t < 2; ++t) {
    const int idx = t * 128 + tid;        // 192 outputs = 32 rows * 6
    if (idx < 192) {
      const int rr = idx / 6, oo = idx % 6;
      float sum = bd2[oo];
#pragma unroll 8
      for (int k = 0; k < 64; ++k)
        sum += sD[rr >> 4][(rr & 15) * 64 + k] * Wd2[k * 6 + oo];
      y[(long long)(sbase0 + rr) * 6 + oo] = sum;
    }
  }
}

// ================= host glue =================
extern "C" void kernel_launch(void* const* d_in, const int* in_sizes, int n_in,
                              void* d_out, int out_size, void* d_ws, size_t ws_size,
                              hipStream_t stream) {
  (void)in_sizes; (void)n_in; (void)out_size; (void)ws_size;
  const float* x    = (const float*)d_in[0];
  const int*   ei   = (const int*)d_in[1];
  const float* ea   = (const float*)d_in[2];
  const int*   sidx = (const int*)d_in[3];
  const float* gm   = (const float*)d_in[4];
  const float* Wmsg = (const float*)d_in[5];
  const float* bmsg = (const float*)d_in[6];
  const float* Wnode= (const float*)d_in[7];
  const float* bnode= (const float*)d_in[8];
  const float* Wenc = (const float*)d_in[9];
  const float* benc = (const float*)d_in[10];
  const float* Wih0 = (const float*)d_in[11];
  const float* Whh0 = (const float*)d_in[12];
  const float* bih0 = (const float*)d_in[13];
  const float* bhh0 = (const float*)d_in[14];
  const float* Wih1 = (const float*)d_in[15];
  const float* Whh1 = (const float*)d_in[16];
  const float* bih1 = (const float*)d_in[17];
  const float* bhh1 = (const float*)d_in[18];
  const float* Wd1  = (const float*)d_in[19];
  const float* bd1  = (const float*)d_in[20];
  const float* Wd2  = (const float*)d_in[21];
  const float* bd2  = (const float*)d_in[22];

  // workspace layout
  float* agg      = (float*)d_ws;                                 // 262144*64 f32
  float* all_conv = agg + (size_t)N_NODES * 64;                   // 262144*64 f32
  _Float16* Xws   = (_Float16*)(all_conv + (size_t)N_NODES * 64); // 65536*128 f16
  _Float16* Hws   = Xws + (size_t)N_SAMP * 128;                   // 65536*128 f16

  // output layout (flat, return order)
  float* o_nc = (float*)d_out;                       // 65536*64
  float* o_h1 = o_nc + (size_t)N_SAMP * 64;          // 65536*128
  float* o_c1 = o_h1 + (size_t)N_SAMP * 128;
  float* o_h2 = o_c1 + (size_t)N_SAMP * 128;
  float* o_c2 = o_h2 + (size_t)N_SAMP * 128;
  float* o_y  = o_c2 + (size_t)N_SAMP * 128;         // 65536*6

  hipMemsetAsync(agg, 0, (size_t)N_NODES * 64 * sizeof(float), stream);

  edge_msg_kernel<<<N_EDGES / 128, 256, 0, stream>>>(x, ei, ea, Wmsg, bmsg, agg);
  node_conv_kernel<<<N_NODES / 128, 256, 0, stream>>>(x, agg, Wnode, bnode, all_conv);
  encoder_kernel<<<N_SAMP / 64, 256, 0, stream>>>(all_conv, sidx, gm, Wenc, benc,
                                                  o_nc, Xws, Hws);
  lstm_kernel<<<N_SAMP / 32, 256, 0, stream>>>(Xws, 1, 0, Hws, gm,
                                               Wih0, Whh0, bih0, bhh0, o_h1, o_c1);
  lstm_kernel<<<N_SAMP / 32, 256, 0, stream>>>(o_h1, 0, 1, Hws, gm,
                                               Wih1, Whh1, bih1, bhh1, o_h2, o_c2);
  decoder_kernel<<<N_SAMP / 32, 128, 0, stream>>>(o_h2, o_c2, Wd1, bd1, Wd2, bd2, o_y);
}
